// Transformer_29480655520316
// MI455X (gfx1250) — compile-verified, matrix operand order checked by reading, not run on hardware
//
#include <hip/hip_runtime.h>
#include <math.h>

typedef __attribute__((ext_vector_type(16))) __bf16 v16bf;
typedef __attribute__((ext_vector_type(8)))  float  v8f;
typedef __attribute__((ext_vector_type(4)))  unsigned int v4u;
typedef __attribute__((ext_vector_type(8)))  int  v8i;
typedef __attribute__((ext_vector_type(4)))  int  v4i;

#define LAYERS 4
#define NH     16
#define DM     1024
#define HDIM   64
#define VOCAB  32000
#define SEQ    1024
#define BATCH  2
#define MROWS  (BATCH * SEQ)   // 2048

// ---- dynamic-LDS layout for the GEMM kernel (no static LDS => offsets start at 0,
//      which lets us feed raw byte offsets to the TDM descriptor's lds_addr field) ----
#define ARAW_OFF   0                         // 2 x (128*32) f32  = 32768 B
#define ARAW_BYTES (128 * 32 * 4)
#define BRAW_OFF   (2 * ARAW_BYTES)          // 2 x (32*64) f32   = 16384 B
#define BRAW_BYTES (32 * 64 * 4)
#define AS_OFF     (BRAW_OFF + 2 * BRAW_BYTES)   // 128*34 bf16 = 8704 B
#define BS_OFF     (AS_OFF + 128 * 34 * 2)       // 64*34 bf16  = 4352 B
#define SMEM_BYTES (BS_OFF + 64 * 34 * 2)        // 62208 B total

// K-pair base index for 16-bit WMMA A/B operands (ISA 7.12.2):
// VGPR i holds K = {2i, 2i+1} for i<4, {16+2(i-4), ...} for i>=4; lanes 16-31 are +8.
__device__ __forceinline__ int KB(int i, int kh) {
  return ((i < 4) ? (i * 2) : (16 + (i - 4) * 2)) + kh * 8;
}

__device__ __forceinline__ v8f wmma_bf16(v16bf a, v16bf b, v8f c) {
  return __builtin_amdgcn_wmma_f32_16x16x32_bf16(false, a, false, b, (short)0, c, false, false);
}

// Issue a TDM 2-D tile load: rows x width fp32 elements, row stride (elements),
// landing contiguously at LDS byte offset lds_off. D# per CDNA5 ISA §8.3/8.4.
// This toolchain's builtin takes 6 args (g0, g1, g2, g3, g_extra, cpol).
__device__ __forceinline__ void tdm_load_2d(unsigned lds_off, const float* gptr,
                                            int width, int rows, long long row_stride) {
  unsigned long long ga = (unsigned long long)(__SIZE_TYPE__)gptr;
  v4u g0;
  g0[0] = 1u;                                            // count=1, user mode, no gather
  g0[1] = lds_off;                                       // lds_addr
  g0[2] = (unsigned)(ga & 0xFFFFFFFFu);                  // global_addr[31:0]
  g0[3] = (unsigned)((ga >> 32) & 0x1FFFFFFu) | (2u << 30); // addr[56:32] | type=2
  unsigned td0 = (unsigned)row_stride;                   // tensor_dim0 (in-bounds tiles)
  unsigned td1 = 0x40000000u;                            // tensor_dim1: huge, no OOB clip
  unsigned long long st0 = (unsigned long long)row_stride;
  v8i g1;
  g1[0] = (int)(2u << 16);                               // wg_mask=0, data_size=2 (4B)
  g1[1] = (int)((td0 & 0xFFFFu) << 16);                  // tensor_dim0[15:0] @ bits63:48
  g1[2] = (int)(((td0 >> 16) & 0xFFFFu) | ((td1 & 0xFFFFu) << 16));
  g1[3] = (int)(((td1 >> 16) & 0xFFFFu) | ((unsigned)width << 16));  // tile_dim0
  g1[4] = (int)((unsigned)rows & 0xFFFFu);               // tile_dim1 (tile_dim2=0)
  g1[5] = (int)(st0 & 0xFFFFFFFFu);                      // tensor_dim0_stride[31:0]
  g1[6] = (int)((st0 >> 32) & 0xFFFFu);                  // stride[47:32], dim1_stride=0
  g1[7] = 0;
  v4i z4 = {0, 0, 0, 0};                                 // 2-D tensor: groups 2/3 unused
  v8i z8 = {0, 0, 0, 0, 0, 0, 0, 0};
  __builtin_amdgcn_tensor_load_to_lds(g0, g1, z4, z4, z8, 0);
}

// ---------------- embedding + positional ----------------
__global__ __launch_bounds__(256) void embed_kernel(const int* __restrict__ ids,
                                                    const float* __restrict__ emb,
                                                    const float* __restrict__ pos,
                                                    float* __restrict__ x) {
  int row = blockIdx.x;
  int s   = row % SEQ;
  int id  = ids[row];
  int i0  = threadIdx.x * 4;
  const float* e = emb + (size_t)id * DM;
  const float* p = pos + (size_t)s  * DM;
  float* xr = x + (size_t)row * DM;
  #pragma unroll
  for (int j = 0; j < 4; ++j) xr[i0 + j] = e[i0 + j] + p[i0 + j];
}

// ---------------- layernorm ----------------
__global__ __launch_bounds__(256) void ln_kernel(const float* __restrict__ x,
                                                 const float* __restrict__ g,
                                                 const float* __restrict__ bta,
                                                 float* __restrict__ out) {
  __shared__ float redS[8], redQ[8];
  int row = blockIdx.x;
  const float* xr = x + (size_t)row * DM;
  int i0 = threadIdx.x * 4;
  float4 v = *((const float4*)(xr + i0));
  float s = v.x + v.y + v.z + v.w;
  float q = v.x * v.x + v.y * v.y + v.z * v.z + v.w * v.w;
  #pragma unroll
  for (int off = 16; off; off >>= 1) {
    s += __shfl_xor(s, off, 32);
    q += __shfl_xor(q, off, 32);
  }
  int wid = threadIdx.x >> 5;
  if ((threadIdx.x & 31) == 0) { redS[wid] = s; redQ[wid] = q; }
  __syncthreads();
  float ts = 0.f, tq = 0.f;
  #pragma unroll
  for (int i = 0; i < 8; ++i) { ts += redS[i]; tq += redQ[i]; }
  float mu  = ts * (1.0f / DM);
  float var = tq * (1.0f / DM) - mu * mu;
  float rstd = rsqrtf(var + 1e-5f);
  float* orow = out + (size_t)row * DM;
  #pragma unroll
  for (int j = 0; j < 4; ++j) {
    float xv = (j == 0 ? v.x : j == 1 ? v.y : j == 2 ? v.z : v.w);
    orow[i0 + j] = (xv - mu) * rstd * g[i0 + j] + bta[i0 + j];
  }
}

// ---------------- TDM + WMMA GEMM ----------------
// C[M,N] = act(A[M,K] @ B[K,N] + bias[N]) (+ res[M,N])
// BMODE 0: B row-major, ldb. BMODE 1: B is [H, K, 64] (Wq/Wk/Wv); a 64-wide N tile
// spans exactly one head => plain 2-D tile with row stride 64.
template <int BMODE, bool GELU, bool RESID>
__global__ __launch_bounds__(256) void gemm_kernel(const float* __restrict__ A,
                                                   const float* __restrict__ Bp,
                                                   const float* __restrict__ bias,
                                                   const float* __restrict__ res,
                                                   float* __restrict__ C,
                                                   int M, int N, int K, int ldb) {
  extern __shared__ char smem[];
  float* Araw = (float*)(smem + ARAW_OFF);          // [2][128*32]
  float* Braw = (float*)(smem + BRAW_OFF);          // [2][32*64]
  __bf16 (*As)[34] = (__bf16(*)[34])(smem + AS_OFF);  // [128][34] padded
  __bf16 (*Bs)[34] = (__bf16(*)[34])(smem + BS_OFF);  // [64][34]  padded, B transposed

  const int tid  = threadIdx.x;
  const int lane = tid & 31;
  const int kh   = lane >> 4;
  const int cidx = lane & 15;
  const int w    = tid >> 5;
  const int wm   = w & 3;
  const int wn   = w >> 2;
  const int bm   = blockIdx.y * 128;
  const int bn   = blockIdx.x * 64;

  const float* Abase = A + (size_t)bm * K;
  const float* Bbase = (BMODE == 0) ? (Bp + bn) : (Bp + (size_t)(bn >> 6) * K * 64);
  const long long bstride = (BMODE == 0) ? (long long)ldb : 64ll;

  v8f acc[2][2];
  #pragma unroll
  for (int ms = 0; ms < 2; ++ms)
    #pragma unroll
    for (int nt = 0; nt < 2; ++nt)
      #pragma unroll
      for (int r = 0; r < 8; ++r) acc[ms][nt][r] = 0.f;

  const int ksteps = K >> 5;
  if (tid < 32) {  // wave 0 drives the TDM pipeline
    tdm_load_2d(ARAW_OFF, Abase, 32, 128, (long long)K);
    tdm_load_2d(BRAW_OFF, Bbase + 0, 64, 32, bstride);
  }

  for (int s = 0; s < ksteps; ++s) {
    const int cur = s & 1;
    if (tid < 32) {
      if (s + 1 < ksteps) {  // prefetch next K-step into the other buffer
        int k1 = (s + 1) << 5;
        tdm_load_2d(ARAW_OFF + (cur ^ 1) * ARAW_BYTES, Abase + k1, 32, 128, (long long)K);
        tdm_load_2d(BRAW_OFF + (cur ^ 1) * BRAW_BYTES,
                    Bbase + (size_t)k1 * bstride, 64, 32, bstride);
        __builtin_amdgcn_s_wait_tensorcnt(2);  // in-order TDM: current pair has landed
      } else {
        __builtin_amdgcn_s_wait_tensorcnt(0);
      }
    }
    __syncthreads();  // publish TDM-written LDS to all waves

    { // convert fp32 tiles -> bf16 operand tiles (once per element)
      const float* ar = Araw + cur * (128 * 32);
      int row = tid >> 1;
      int cb  = (tid & 1) * 16;
      #pragma unroll
      for (int j = 0; j < 16; ++j) As[row][cb + j] = (__bf16)ar[row * 32 + cb + j];
      const float* br = Braw + cur * (32 * 64);
      int kk  = tid >> 3;
      int cb2 = (tid & 7) * 8;
      #pragma unroll
      for (int j = 0; j < 8; ++j) Bs[cb2 + j][kk] = (__bf16)br[kk * 64 + cb2 + j];
    }
    __syncthreads();

    v16bf a[2];
    #pragma unroll
    for (int ms = 0; ms < 2; ++ms) {
      int m = wm * 32 + ms * 16 + cidx;
      #pragma unroll
      for (int i = 0; i < 8; ++i) {
        int kb = KB(i, kh);
        a[ms][2 * i]     = As[m][kb];
        a[ms][2 * i + 1] = As[m][kb + 1];
      }
    }
    #pragma unroll
    for (int nt = 0; nt < 2; ++nt) {
      int n = wn * 32 + nt * 16 + cidx;
      v16bf bb;
      #pragma unroll
      for (int i = 0; i < 8; ++i) {
        int kb = KB(i, kh);
        bb[2 * i]     = Bs[n][kb];
        bb[2 * i + 1] = Bs[n][kb + 1];
      }
      #pragma unroll
      for (int ms = 0; ms < 2; ++ms)
        acc[ms][nt] = wmma_bf16(a[ms], bb, acc[ms][nt]);
    }
    __syncthreads();  // As/Bs consumed before next convert overwrites them
  }

  #pragma unroll
  for (int ms = 0; ms < 2; ++ms)
    #pragma unroll
    for (int nt = 0; nt < 2; ++nt)
      #pragma unroll
      for (int r = 0; r < 8; ++r) {
        int m = bm + wm * 32 + ms * 16 + r + 8 * kh;
        int n = bn + wn * 32 + nt * 16 + cidx;
        float vv = acc[ms][nt][r] + bias[n];
        if (GELU)  vv = 0.5f * vv * (1.0f + erff(vv * 0.70710678118654752f));
        if (RESID) vv += res[(size_t)m * N + n];
        C[(size_t)m * N + n] = vv;
      }
}

// ---------------- flash attention, one wave per (b,h,16 query rows) ----------------
__global__ __launch_bounds__(32) void attn_kernel(const float* __restrict__ Q,
                                                  const float* __restrict__ Kt,
                                                  const float* __restrict__ Vt,
                                                  float* __restrict__ X) {
  const int lane = threadIdx.x;
  const int kh   = lane >> 4;
  const int cidx = lane & 15;
  const int m0   = blockIdx.x * 16;
  const int bh   = blockIdx.y;
  const int b    = bh / NH;
  const int h    = bh % NH;

  const float* qbase = Q  + ((size_t)b * SEQ) * DM + h * HDIM;
  const float* kbase = Kt + ((size_t)b * SEQ) * DM + h * HDIM;
  const float* vbase = Vt + ((size_t)b * SEQ) * DM + h * HDIM;

  __shared__ __bf16 Qs[16][72];
  __shared__ __bf16 Ps[16][34];

  #pragma unroll
  for (int j = 0; j < 32; ++j) {
    int idx = j * 32 + lane;
    int m = idx >> 6, kk = idx & 63;
    Qs[m][kk] = (__bf16)qbase[(size_t)(m0 + m) * DM + kk];
  }
  __syncthreads();

  v16bf a_lo, a_hi;
  #pragma unroll
  for (int i = 0; i < 8; ++i) {
    int kb = KB(i, kh);
    a_lo[2 * i]     = Qs[cidx][kb];
    a_lo[2 * i + 1] = Qs[cidx][kb + 1];
    a_hi[2 * i]     = Qs[cidx][kb + 32];
    a_hi[2 * i + 1] = Qs[cidx][kb + 33];
  }

  float mi[8], li[8];
  v8f o[4];
  #pragma unroll
  for (int r = 0; r < 8; ++r) { mi[r] = -1e30f; li[r] = 0.f; }
  #pragma unroll
  for (int nt = 0; nt < 4; ++nt)
    #pragma unroll
    for (int r = 0; r < 8; ++r) o[nt][r] = 0.f;

  const int tend = m0 + 16;
  for (int t0 = 0; t0 < tend; t0 += 32) {
    v8f s0, s1;
    #pragma unroll
    for (int r = 0; r < 8; ++r) { s0[r] = 0.f; s1[r] = 0.f; }
    {
      int t = t0 + cidx;
      const float* kp = kbase + (size_t)t * DM;
      v16bf b0, b1;
      #pragma unroll
      for (int i = 0; i < 8; ++i) {
        int kb = KB(i, kh);
        b0[2 * i]     = (__bf16)kp[kb];
        b0[2 * i + 1] = (__bf16)kp[kb + 1];
        b1[2 * i]     = (__bf16)kp[kb + 32];
        b1[2 * i + 1] = (__bf16)kp[kb + 33];
      }
      s0 = wmma_bf16(a_lo, b0, s0);
      s0 = wmma_bf16(a_hi, b1, s0);
    }
    {
      int t = t0 + 16 + cidx;
      int tc = (t < SEQ) ? t : (SEQ - 1);
      const float* kp = kbase + (size_t)tc * DM;
      v16bf b0, b1;
      #pragma unroll
      for (int i = 0; i < 8; ++i) {
        int kb = KB(i, kh);
        b0[2 * i]     = (__bf16)kp[kb];
        b0[2 * i + 1] = (__bf16)kp[kb + 1];
        b1[2 * i]     = (__bf16)kp[kb + 32];
        b1[2 * i + 1] = (__bf16)kp[kb + 33];
      }
      s1 = wmma_bf16(a_lo, b0, s1);
      s1 = wmma_bf16(a_hi, b1, s1);
    }

    float corr[8];
    #pragma unroll
    for (int r = 0; r < 8; ++r) {
      int m = m0 + r + 8 * kh;
      float v0 = s0[r] * 0.125f;
      float v1 = s1[r] * 0.125f;
      if (t0 + cidx      > m) v0 = -1e30f;
      if (t0 + 16 + cidx > m) v1 = -1e30f;
      float bmax = fmaxf(v0, v1);
      #pragma unroll
      for (int off = 8; off; off >>= 1) bmax = fmaxf(bmax, __shfl_xor(bmax, off, 32));
      float mnew = fmaxf(mi[r], bmax);
      corr[r] = __expf(mi[r] - mnew);
      float p0 = __expf(v0 - mnew);
      float p1 = __expf(v1 - mnew);
      float rs = p0 + p1;
      #pragma unroll
      for (int off = 8; off; off >>= 1) rs += __shfl_xor(rs, off, 32);
      li[r] = li[r] * corr[r] + rs;
      mi[r] = mnew;
      Ps[r + 8 * kh][cidx]      = (__bf16)p0;
      Ps[r + 8 * kh][16 + cidx] = (__bf16)p1;
    }
    #pragma unroll
    for (int nt = 0; nt < 4; ++nt)
      #pragma unroll
      for (int r = 0; r < 8; ++r) o[nt][r] *= corr[r];
    __syncthreads();

    v16bf pa;
    #pragma unroll
    for (int i = 0; i < 8; ++i) {
      int kb = KB(i, kh);
      pa[2 * i]     = Ps[cidx][kb];
      pa[2 * i + 1] = Ps[cidx][kb + 1];
    }
    #pragma unroll
    for (int nt = 0; nt < 4; ++nt) {
      v16bf bv;
      #pragma unroll
      for (int i = 0; i < 8; ++i) {
        int kb = KB(i, kh);
        int t1 = t0 + kb;     if (t1 >= tend) t1 = tend - 1;
        int t2 = t0 + kb + 1; if (t2 >= tend) t2 = tend - 1;
        bv[2 * i]     = (__bf16)vbase[(size_t)t1 * DM + nt * 16 + cidx];
        bv[2 * i + 1] = (__bf16)vbase[(size_t)t2 * DM + nt * 16 + cidx];
      }
      o[nt] = wmma_bf16(pa, bv, o[nt]);
    }
    __syncthreads();
  }

  #pragma unroll
  for (int r = 0; r < 8; ++r) {
    float inv = 1.0f / li[r];
    size_t row = (size_t)(b * SEQ + m0 + r + 8 * kh);
    float* xr = X + row * DM + h * HDIM;
    #pragma unroll
    for (int nt = 0; nt < 4; ++nt)
      xr[nt * 16 + cidx] += o[nt][r] * inv;
  }
}

// ---------------- host orchestration ----------------
extern "C" void kernel_launch(void* const* d_in, const int* in_sizes, int n_in,
                              void* d_out, int out_size, void* d_ws, size_t ws_size,
                              hipStream_t stream) {
  (void)in_sizes; (void)n_in; (void)out_size; (void)ws_size;
  const int*   ids  = (const int*)  d_in[0];
  const float* emb  = (const float*)d_in[1];
  const float* pos  = (const float*)d_in[2];
  const float* Wq   = (const float*)d_in[3];
  const float* bq   = (const float*)d_in[4];
  const float* Wk   = (const float*)d_in[5];
  const float* bk   = (const float*)d_in[6];
  const float* Wv   = (const float*)d_in[7];
  const float* bv   = (const float*)d_in[8];
  const float* ln1g = (const float*)d_in[9];
  const float* ln1b = (const float*)d_in[10];
  const float* ln2g = (const float*)d_in[11];
  const float* ln2b = (const float*)d_in[12];
  const float* W1   = (const float*)d_in[13];
  const float* b1   = (const float*)d_in[14];
  const float* W2   = (const float*)d_in[15];
  const float* b2   = (const float*)d_in[16];
  const float* pw   = (const float*)d_in[17];
  const float* pb   = (const float*)d_in[18];
  float* out = (float*)d_out;

  const size_t MD = (size_t)MROWS * DM;       // 2M floats
  float* ws   = (float*)d_ws;
  float* xb   = ws;                           // [B,S,D]
  float* hb   = ws + MD;                      // [B,S,D]
  float* qb   = ws + 2 * MD;                  // [B,S,H,HD]
  float* kbuf = ws + 3 * MD;
  float* vbuf = ws + 4 * MD;
  float* ffb  = ws + 5 * MD;                  // [B,S,4D]

  embed_kernel<<<MROWS, 256, 0, stream>>>(ids, emb, pos, xb);

  for (int l = 0; l < LAYERS; ++l) {
    const size_t wqo = (size_t)l * NH * DM * HDIM;
    const size_t w1o = (size_t)l * DM * 4 * DM;

    ln_kernel<<<MROWS, 256, 0, stream>>>(xb, ln1g + l * DM, ln1b + l * DM, hb);

    dim3 gqkv(DM / 64, MROWS / 128);
    gemm_kernel<1, false, false><<<gqkv, 256, SMEM_BYTES, stream>>>(hb, Wq + wqo, bq + l * DM, nullptr, qb,   MROWS, DM, DM, DM);
    gemm_kernel<1, false, false><<<gqkv, 256, SMEM_BYTES, stream>>>(hb, Wk + wqo, bk + l * DM, nullptr, kbuf, MROWS, DM, DM, DM);
    gemm_kernel<1, false, false><<<gqkv, 256, SMEM_BYTES, stream>>>(hb, Wv + wqo, bv + l * DM, nullptr, vbuf, MROWS, DM, DM, DM);

    attn_kernel<<<dim3(SEQ / 16, BATCH * NH), 32, 0, stream>>>(qb, kbuf, vbuf, xb);

    ln_kernel<<<MROWS, 256, 0, stream>>>(xb, ln2g + l * DM, ln2b + l * DM, hb);

    gemm_kernel<0, true, false><<<dim3(4 * DM / 64, MROWS / 128), 256, SMEM_BYTES, stream>>>(
        hb, W1 + w1o, b1 + (size_t)l * 4 * DM, nullptr, ffb, MROWS, 4 * DM, DM, 4 * DM);
    gemm_kernel<0, false, true><<<dim3(DM / 64, MROWS / 128), 256, SMEM_BYTES, stream>>>(
        ffb, W2 + w1o, b2 + l * DM, xb, xb, MROWS, DM, 4 * DM, DM);
  }

  gemm_kernel<0, false, false><<<dim3(VOCAB / 64, MROWS / 128), 256, SMEM_BYTES, stream>>>(
      xb, pw, pb, nullptr, out, MROWS, VOCAB, DM, VOCAB);
}
